// RNN_65360812310967
// MI455X (gfx1250) — matrix operand order repaired and checked by reading, not compiled, not run on hardware
//
#include <hip/hip_runtime.h>
#include <hip/hip_bf16.h>
#include <math.h>

// Problem constants (from reference): T=1024, B=64, D=H=512, L=2
#define T_STEPS 1024
#define BATCH   64
#define DIM     512
#define NWG     4          // persistent workgroups cooperating on the scan

// Scan LDS layout: weight slice + state buffer (padded pitch, bank-conflict free)
#define W_ELEMS   (8 * 16 * 512)          // 65536 bf16 = 128 KB
#define S_PITCH   520                     // 520 elems = 260 dwords == 4 (mod 64)
#define SCAN_LDS  ((W_ELEMS + BATCH * S_PITCH) * 2)   // 197632 B < 320 KB

typedef __attribute__((ext_vector_type(16))) __bf16 v16bf;
typedef __attribute__((ext_vector_type(8)))  float  v8f;

union FragBF { uint4 u[2]; v16bf v; };

__device__ __forceinline__ unsigned short f2bf(float f) {
    unsigned u = __float_as_uint(f);
    u += 0x7FFFu + ((u >> 16) & 1u);          // round-to-nearest-even
    return (unsigned short)(u >> 16);
}
__device__ __forceinline__ unsigned pk2(float lo, float hi) {
    return (unsigned)f2bf(lo) | ((unsigned)f2bf(hi) << 16);
}

// ---------------------------------------------------------------------------
// Elementwise fp32 -> bf16 (8 elems/thread), bandwidth bound.
// ---------------------------------------------------------------------------
__global__ void cvt_bf16_8(const float* __restrict__ in, unsigned short* __restrict__ out) {
    size_t i = ((size_t)blockIdx.x * blockDim.x + threadIdx.x) * 8;
    float4 f0 = *(const float4*)(in + i);
    float4 f1 = *(const float4*)(in + i + 4);
    *(uint4*)(out + i) = make_uint4(pk2(f0.x, f0.y), pk2(f0.z, f0.w),
                                    pk2(f1.x, f1.y), pk2(f1.z, f1.w));
}

// ---------------------------------------------------------------------------
// Pre-swizzle a 512x512 fp32 weight into this file's B-fragment order (contiguous k, as A):
// tile (nt,kt): lane l holds n = nt*16 + (l&15), k = kt*32 + (l>>4)*16 + j.
// ---------------------------------------------------------------------------
__global__ void swizzle_w(const float* __restrict__ w, unsigned short* __restrict__ out) {
    int idx = blockIdx.x * blockDim.x + threadIdx.x;   // 0..262143
    int j  =  idx        & 15;
    int l  = (idx >> 4)  & 31;
    int kt = (idx >> 9)  & 15;
    int nt =  idx >> 13;                                // 0..31
    int n  = nt * 16 + (l & 15);
    int k  = kt * 32 + (l >> 4) * 16 + j;
    out[idx] = f2bf(w[n * DIM + k]);
}

// Reset the inter-WG barrier and load the initial recurrent state (bf16).
__global__ void init_scan(const float* __restrict__ s0,
                          unsigned short* __restrict__ scomm,
                          unsigned* __restrict__ bar) {
    int i = blockIdx.x * blockDim.x + threadIdx.x;
    if (i == 0) *bar = 0u;
    if (i < BATCH * DIM) scomm[i] = f2bf(s0[i]);
}

// ---------------------------------------------------------------------------
// xproj = A @ W^T, A = [65536 x 512] bf16, W pre-swizzled bf16 staged in LDS.
// grid = (8, 512), block = 256 (8 waves). Wave: 1 M-tile x 4 N-tiles.
// Both A (global) and B (LDS) fragments are software-pipelined one k-step
// ahead so no WMMA waits on the loads feeding it.
// ---------------------------------------------------------------------------
__global__ void xproj_gemm(const unsigned short* __restrict__ Ab,
                           const unsigned short* __restrict__ Bs,
                           float* __restrict__ C) {
    extern __shared__ unsigned short ldsb[];           // 4*16*512 bf16 = 64 KB
    const int nt0 = blockIdx.x * 4;
    {   // stage this block's B slice (4 n_tiles x 16 k_tiles) into LDS
        const uint4* src = (const uint4*)(Bs + (size_t)nt0 * 16 * 512);
        uint4* dst = (uint4*)ldsb;
        #pragma unroll
        for (int i = threadIdx.x; i < 4096; i += 256) dst[i] = src[i];
    }
    __syncthreads();

    const int wave = threadIdx.x >> 5;
    const int lane = threadIdx.x & 31;
    const int half = lane >> 4;
    const int lm   = lane & 15;
    const int mt   = blockIdx.y * 8 + wave;
    const int mrow = mt * 16 + lm;

    v8f zero = {0.f,0.f,0.f,0.f,0.f,0.f,0.f,0.f};
    v8f c[4] = {zero, zero, zero, zero};

    const unsigned short* abase = Ab + (size_t)mrow * DIM + half * 16;
    const unsigned short* bbase = ldsb + lane * 16;

    FragBF an, bn[4];                            // prefetched fragments (kt+1)
    an.u[0] = *(const uint4*)(abase);
    an.u[1] = *(const uint4*)(abase + 8);
    #pragma unroll
    for (int q = 0; q < 4; ++q) {
        const unsigned short* bp = bbase + (q * 16) * 512;
        bn[q].u[0] = *(const uint4*)(bp);
        bn[q].u[1] = *(const uint4*)(bp + 8);
    }

    #pragma unroll
    for (int kt = 0; kt < 16; ++kt) {
        FragBF a = an;
        FragBF b[4] = {bn[0], bn[1], bn[2], bn[3]};
        if (kt < 15) {                           // prefetch next k-step
            const unsigned short* ap = abase + (kt + 1) * 32;
            an.u[0] = *(const uint4*)(ap);
            an.u[1] = *(const uint4*)(ap + 8);
            #pragma unroll
            for (int q = 0; q < 4; ++q) {
                const unsigned short* bp = bbase + (q * 16 + kt + 1) * 512;
                bn[q].u[0] = *(const uint4*)(bp);
                bn[q].u[1] = *(const uint4*)(bp + 8);
            }
        }
        #pragma unroll
        for (int q = 0; q < 4; ++q)
            c[q] = __builtin_amdgcn_wmma_f32_16x16x32_bf16(
                false, a.v, false, b[q].v, (short)0, c[q], false, false);
    }

    #pragma unroll
    for (int q = 0; q < 4; ++q) {
        int n = (nt0 + q) * 16 + lm;
        #pragma unroll
        for (int r = 0; r < 8; ++r) {
            int m = mt * 16 + r + half * 8;
            C[(size_t)m * DIM + n] = c[q][r];
        }
    }
}

// ---------------------------------------------------------------------------
// Recurrent scan: 4 persistent WGs, each owns 128 output columns; bf16 weight
// slice resident in LDS. Per step: stage state (64 KB) into padded LDS,
// s_new = tanh(xproj_t + s @ wt^T + b). State double-buffered in global,
// one 4-WG atomic barrier per step. block = 512 (16 waves).
// ---------------------------------------------------------------------------
template<bool LAST_LAYER>
__global__ void rnn_scan(const float* __restrict__ p,
                         const unsigned short* __restrict__ wts,
                         const float* __restrict__ bt,
                         unsigned short* __restrict__ scomm,
                         unsigned* __restrict__ bar,
                         unsigned short* __restrict__ hob,
                         float* __restrict__ hof,
                         float* __restrict__ sfinal) {
    extern __shared__ unsigned short lds[];
    unsigned short* lds_w = lds;                 // 65536 elems (128 KB)
    unsigned short* lds_s = lds + W_ELEMS;       // 64 x 520 elems (padded state)
    const int wg = blockIdx.x;

    {   // stage this WG's weight slice (n_tiles wg*8 .. wg*8+7) into LDS
        const uint4* src = (const uint4*)(wts + (size_t)wg * W_ELEMS);
        uint4* dst = (uint4*)lds_w;
        #pragma unroll
        for (int i = threadIdx.x; i < W_ELEMS / 8; i += 512) dst[i] = src[i];
    }

    const int wave = threadIdx.x >> 5;
    const int lane = threadIdx.x & 31;
    const int half = lane >> 4;
    const int lm   = lane & 15;
    const int mt   = wave & 3;            // 4 M-tiles cover B=64
    const int np   = wave >> 2;           // 4 groups x 2 n_tiles = 8 local tiles
    const int nl0  = np * 2, nl1 = np * 2 + 1;
    const int n0   = (wg * 8 + nl0) * 16 + lm;
    const int n1   = (wg * 8 + nl1) * 16 + lm;
    const float bias0 = bt[n0];
    const float bias1 = bt[n1];
    const unsigned short* wb0 = lds_w + (nl0 * 16) * 512 + lane * 16;
    const unsigned short* wb1 = lds_w + (nl1 * 16) * 512 + lane * 16;
    __syncthreads();

    for (int t = 0; t < T_STEPS; ++t) {
        const unsigned short* sin  = scomm + (t & 1)       * (BATCH * DIM);
        unsigned short*       sout = scomm + ((t + 1) & 1) * (BATCH * DIM);
        const float*          prow = p + (size_t)t * (BATCH * DIM);

        // Prefetch epilogue xproj fragment early (overlaps staging + WMMAs).
        float xp0[8], xp1[8];
        #pragma unroll
        for (int r = 0; r < 8; ++r) {
            int m = mt * 16 + r + half * 8;
            xp0[r] = prow[m * DIM + n0];
            xp1[r] = prow[m * DIM + n1];
        }

        {   // stage state into LDS with padded pitch (bank-conflict-free A reads)
            const uint4* src = (const uint4*)sin;
            #pragma unroll
            for (int i = threadIdx.x; i < 4096; i += 512) {
                int row = i >> 6, c16 = i & 63;
                *(uint4*)(lds_s + row * S_PITCH + c16 * 8) = src[i];
            }
        }
        __syncthreads();

        v8f c0 = {0.f,0.f,0.f,0.f,0.f,0.f,0.f,0.f};
        v8f c1 = {0.f,0.f,0.f,0.f,0.f,0.f,0.f,0.f};

        const unsigned short* arow = lds_s + (mt * 16 + lm) * S_PITCH + half * 16;

        FragBF an, b0n, b1n;                     // prefetched fragments
        an.u[0]  = *(const uint4*)(arow);
        an.u[1]  = *(const uint4*)(arow + 8);
        b0n.u[0] = *(const uint4*)(wb0);
        b0n.u[1] = *(const uint4*)(wb0 + 8);
        b1n.u[0] = *(const uint4*)(wb1);
        b1n.u[1] = *(const uint4*)(wb1 + 8);

        #pragma unroll
        for (int kt = 0; kt < 16; ++kt) {
            FragBF a = an, b0 = b0n, b1 = b1n;
            if (kt < 15) {                       // prefetch next k-step
                const unsigned short* ap  = arow + (kt + 1) * 32;
                const unsigned short* bp0 = wb0 + (kt + 1) * 512;
                const unsigned short* bp1 = wb1 + (kt + 1) * 512;
                an.u[0]  = *(const uint4*)(ap);
                an.u[1]  = *(const uint4*)(ap + 8);
                b0n.u[0] = *(const uint4*)(bp0);
                b0n.u[1] = *(const uint4*)(bp0 + 8);
                b1n.u[0] = *(const uint4*)(bp1);
                b1n.u[1] = *(const uint4*)(bp1 + 8);
            }
            c0 = __builtin_amdgcn_wmma_f32_16x16x32_bf16(
                false, a.v, false, b0.v, (short)0, c0, false, false);
            c1 = __builtin_amdgcn_wmma_f32_16x16x32_bf16(
                false, a.v, false, b1.v, (short)0, c1, false, false);
        }

        #pragma unroll
        for (int r = 0; r < 8; ++r) {
            int m = mt * 16 + r + half * 8;
            float v0 = tanhf(c0[r] + xp0[r] + bias0);
            float v1 = tanhf(c1[r] + xp1[r] + bias1);
            unsigned short q0 = f2bf(v0), q1 = f2bf(v1);
            sout[m * DIM + n0] = q0;
            sout[m * DIM + n1] = q1;
            if (LAST_LAYER) {
                hof[((size_t)t * BATCH + m) * DIM + n0] = v0;
                hof[((size_t)t * BATCH + m) * DIM + n1] = v1;
            } else {
                hob[((size_t)t * BATCH + m) * DIM + n0] = q0;
                hob[((size_t)t * BATCH + m) * DIM + n1] = q1;
            }
            if (t == T_STEPS - 1) {
                sfinal[m * DIM + n0] = v0;
                sfinal[m * DIM + n1] = v1;
            }
        }

        // ---- inter-workgroup step barrier (monotonic counter) ----
        __threadfence();                  // make sout stores agent-visible
        __syncthreads();
        if (threadIdx.x == 0) {
            atomicAdd(bar, 1u);
            unsigned target = (unsigned)(t + 1) * NWG;
            while (*(volatile unsigned*)bar < target)
                __builtin_amdgcn_s_sleep(2);
        }
        __syncthreads();
        __threadfence();                  // acquire: see other WGs' state
    }
}

extern "C" void kernel_launch(void* const* d_in, const int* in_sizes, int n_in,
                              void* d_out, int out_size, void* d_ws, size_t ws_size,
                              hipStream_t stream) {
    const float* x   = (const float*)d_in[0];
    const float* st  = (const float*)d_in[1];
    const float* wl0 = (const float*)d_in[2];
    const float* wt0 = (const float*)d_in[3];
    const float* bt0 = (const float*)d_in[4];
    const float* wl1 = (const float*)d_in[5];
    const float* wt1 = (const float*)d_in[6];
    const float* bt1 = (const float*)d_in[7];
    float* out = (float*)d_out;                       // [T,B,H] h + [2,B,H] states

    uint8_t* w = (uint8_t*)d_ws;
    size_t off = 0;
    float* p = (float*)(w + off);                      off += (size_t)T_STEPS * BATCH * DIM * 4; // 128 MB
    unsigned short* h0b  = (unsigned short*)(w + off); off += (size_t)T_STEPS * BATCH * DIM * 2; // 64 MB (x_bf16, then h0)
    unsigned short* wl0s = (unsigned short*)(w + off); off += (size_t)DIM * DIM * 2;
    unsigned short* wt0s = (unsigned short*)(w + off); off += (size_t)DIM * DIM * 2;
    unsigned short* wl1s = (unsigned short*)(w + off); off += (size_t)DIM * DIM * 2;
    unsigned short* wt1s = (unsigned short*)(w + off); off += (size_t)DIM * DIM * 2;
    unsigned short* scomm = (unsigned short*)(w + off); off += (size_t)2 * BATCH * DIM * 2;
    unsigned* bar = (unsigned*)(w + off);              off += 256;

    float* sf0 = out + (size_t)T_STEPS * BATCH * DIM;
    float* sf1 = sf0 + (size_t)BATCH * DIM;

    // x -> bf16 (staged in the h0b region; consumed by gemm0, then overwritten by scan0)
    cvt_bf16_8<<<dim3((T_STEPS * BATCH * DIM) / (256 * 8)), 256, 0, stream>>>(x, h0b);

    dim3 swzGrid((DIM * DIM) / 256);
    swizzle_w<<<swzGrid, 256, 0, stream>>>(wl0, wl0s);
    swizzle_w<<<swzGrid, 256, 0, stream>>>(wt0, wt0s);
    swizzle_w<<<swzGrid, 256, 0, stream>>>(wl1, wl1s);
    swizzle_w<<<swzGrid, 256, 0, stream>>>(wt1, wt1s);

    // ---- layer 0 ----
    xproj_gemm<<<dim3(8, 512), 256, 64 * 1024, stream>>>(h0b, wl0s, p);
    init_scan<<<dim3(128), 256, 0, stream>>>(st, scomm, bar);
    rnn_scan<false><<<dim3(NWG), 512, SCAN_LDS, stream>>>(
        p, wt0s, bt0, scomm, bar, h0b, nullptr, sf0);

    // ---- layer 1 ----
    xproj_gemm<<<dim3(8, 512), 256, 64 * 1024, stream>>>(h0b, wl1s, p);
    init_scan<<<dim3(128), 256, 0, stream>>>(st + BATCH * DIM, scomm, bar);
    rnn_scan<true><<<dim3(NWG), 512, SCAN_LDS, stream>>>(
        p, wt1s, bt1, scomm, bar, nullptr, out, sf1);
}
